// GATNet_33904471834791
// MI455X (gfx1250) — matrix operand order repaired and checked by reading, not compiled, not run on hardware
//
#include <hip/hip_runtime.h>
#include <hip/hip_bf16.h>
#include <math.h>

// ---------------- model constants ----------------
#define N_NODES 50000
#define N_EDGES 500000
#define BATCH_B 256
#define SEQ     128
#define FXD     78
#define HEADS   10
#define OUTD    128
#define EMBD    128
#define NF      32
#define KCONV   8
#define CONV_L  121          // SEQ - K + 1
#define HC1     (HEADS*FXD)  // 780
#define XTK     (NF*CONV_L)  // 3872
#define VOC     26

typedef __attribute__((ext_vector_type(16))) _Float16 v16h;
typedef __attribute__((ext_vector_type(2)))  _Float16 v2h;
typedef __attribute__((ext_vector_type(8)))  float    v8f;
typedef __attribute__((ext_vector_type(8)))  unsigned v8u;
typedef __attribute__((ext_vector_type(4)))  unsigned u32x4;
typedef __attribute__((ext_vector_type(8)))  int      i32x8;
typedef __attribute__((ext_vector_type(4)))  int      i32x4;

__device__ __forceinline__ v2h cvt_pk_f16(float a, float b) {
    return __builtin_bit_cast(v2h, __builtin_amdgcn_cvt_pkrtz(a, b));
}

// ---------------- helpers ----------------
__device__ __forceinline__ void atomicMaxF(float* addr, float val) {
    if (val >= 0.0f) atomicMax((int*)addr, __float_as_int(val));
    else             atomicMin((unsigned int*)addr, __float_as_uint(val));
}

__global__ void fill_f32(float* __restrict__ p, float v, long long n) {
    long long i = (long long)blockIdx.x * blockDim.x + threadIdx.x;
    long long stride = (long long)gridDim.x * blockDim.x;
    for (; i < n; i += stride) p[i] = v;
}

// ---------------- WMMA GEMM: D = A(f32)[MxK] * B(f32)[KxN] (+bias)(+relu) ----------------
// f16 compute, f32 accumulate. 128 thr = 4 waves; tile 64x64x32; double-buffered LDS.
// Requires: K, N, lda, ldb even (true for every call site) so float2 loads are 8B aligned.
__global__ __launch_bounds__(128) void gemm_wmma_f16(
    const float* __restrict__ A, const float* __restrict__ Bm,
    float* __restrict__ D, const float* __restrict__ bias,
    int M, int N, int K, int lda, int ldb, int ldd, int dcol0, int relu)
{
    __shared__ _Float16 sA[2][64 * 32];    // [m][k]
    __shared__ _Float16 sB[2][64 * 32];    // [n][k] (transposed)
    const int tid   = threadIdx.x;
    const int wave  = tid >> 5;
    const int lane  = tid & 31;
    const int m0    = blockIdx.y * 64;
    const int n0    = blockIdx.x * 64;
    const int mrow  = lane & 15;
    const int khalf = lane >> 4;

    // staging assignment: A row ra, k-seg sa (16 wide); B k-row kb, n-seg nbb (16 wide)
    const int ra  = tid >> 1;
    const int sa  = (tid & 1) * 16;
    const int kb  = tid >> 2;
    const int nbb = (tid & 3) * 16;

    float2 ar[8], br[8];

    auto load_tile = [&](int k0) {
        const int gm  = m0 + ra;
        const int cgm = gm < M - 1 ? gm : M - 1;
        const bool mok = gm < M;
        #pragma unroll
        for (int j = 0; j < 8; ++j) {
            int gk = k0 + sa + 2 * j;
            int ck = gk < K - 2 ? gk : K - 2;          // clamp: load always legal
            float2 v = *reinterpret_cast<const float2*>(&A[(long long)cgm * lda + ck]);
            bool ok = mok && (gk < K);                  // K even, gk even
            ar[j].x = ok ? v.x : 0.0f;
            ar[j].y = ok ? v.y : 0.0f;
        }
        const int gkb = k0 + kb;
        const int ckb = gkb < K - 1 ? gkb : K - 1;
        const bool kok = gkb < K;
        #pragma unroll
        for (int j = 0; j < 8; ++j) {
            int gn = n0 + nbb + 2 * j;
            int cn = gn < N - 2 ? gn : N - 2;
            float2 v = *reinterpret_cast<const float2*>(&Bm[(long long)ckb * ldb + cn]);
            bool ok = kok && (gn < N);                  // N even, gn even
            br[j].x = ok ? v.x : 0.0f;
            br[j].y = ok ? v.y : 0.0f;
        }
    };

    auto store_tile = [&](int buf) {
        v2h* dstA = (v2h*)&sA[buf][ra * 32 + sa];
        #pragma unroll
        for (int j = 0; j < 8; ++j)
            dstA[j] = cvt_pk_f16(ar[j].x, ar[j].y);
        #pragma unroll
        for (int j = 0; j < 8; ++j) {
            v2h h = cvt_pk_f16(br[j].x, br[j].y);
            int n = nbb + 2 * j;
            sB[buf][n * 32 + kb]       = h[0];
            sB[buf][(n + 1) * 32 + kb] = h[1];
        }
    };

    v8f acc[4];
    #pragma unroll
    for (int nt = 0; nt < 4; ++nt)
        #pragma unroll
        for (int r = 0; r < 8; ++r) acc[nt][r] = 0.0f;

    const int ntiles = (K + 31) >> 5;
    load_tile(0);
    store_tile(0);
    __syncthreads();

    for (int kt = 0; kt < ntiles; ++kt) {
        const int cur = kt & 1;
        const bool more = (kt + 1) < ntiles;
        if (more) load_tile((kt + 1) * 32);     // overlap next tile's loads with compute

        // A fragment (ISA layout): lanes 0-15 k in [8h..8h+7] | [16+8h..], m = lane&15
        const unsigned* pA = (const unsigned*)&sA[cur][(wave * 16 + mrow) * 32];
        v8u aw;
        #pragma unroll
        for (int vi = 0; vi < 4; ++vi) aw[vi]     = pA[4 * khalf + vi];
        #pragma unroll
        for (int vi = 0; vi < 4; ++vi) aw[4 + vi] = pA[8 + 4 * khalf + vi];
        v16h afrag = __builtin_bit_cast(v16h, aw);

        // preload ALL B fragments so the 8 ds_load_b128 issue back-to-back,
        // then run the 4 WMMAs as a burst (single dscnt wait instead of 4)
        v8u bw[4];
        #pragma unroll
        for (int nt = 0; nt < 4; ++nt) {
            const unsigned* pB = (const unsigned*)&sB[cur][(nt * 16 + mrow) * 32];
            #pragma unroll
            for (int vi = 0; vi < 8; ++vi) bw[nt][vi] = pB[(vi < 4 ? 0 : 4) + 4 * khalf + vi];
        }
        #pragma unroll
        for (int nt = 0; nt < 4; ++nt) {
            v16h bfrag = __builtin_bit_cast(v16h, bw[nt]);
            acc[nt] = __builtin_amdgcn_wmma_f32_16x16x32_f16(
                false, afrag, false, bfrag, (short)0, acc[nt], false, false);
        }

        if (more) {
            store_tile(cur ^ 1);
            __syncthreads();
        }
    }

    // epilogue: D layout M = r + 8*khalf, N = lane&15
    const int mbase = m0 + wave * 16 + 8 * khalf;
    #pragma unroll
    for (int nt = 0; nt < 4; ++nt) {
        int col = n0 + nt * 16 + mrow;
        if (col >= N) continue;
        float bv = bias ? bias[col] : 0.0f;
        #pragma unroll
        for (int r = 0; r < 8; ++r) {
            int row = mbase + r;
            if (row < M) {
                float val = acc[nt][r] + bv;
                if (relu) val = fmaxf(val, 0.0f);
                D[(long long)row * ldd + dcol0 + col] = val;
            }
        }
    }
}

// ---------------- GAT attention pieces ----------------
__global__ void gat_scores(const float* __restrict__ h, const float* __restrict__ a_src,
                           const float* __restrict__ a_dst, float* __restrict__ ss,
                           float* __restrict__ sd, int Nn, int H, int C)
{
    long long i = (long long)blockIdx.x * blockDim.x + threadIdx.x;
    if (i >= (long long)Nn * H) return;
    int n = (int)(i / H), hh = (int)(i % H);
    const float* hp = h + (long long)n * H * C + (long long)hh * C;
    const float* as = a_src + hh * C;
    const float* ad = a_dst + hh * C;
    float s1 = 0.0f, s2 = 0.0f;
    for (int c = 0; c < C; ++c) { float hv = hp[c]; s1 += hv * as[c]; s2 += hv * ad[c]; }
    ss[i] = s1; sd[i] = s2;
}

__device__ __forceinline__ void edge_sd(const int* ei, long long e, int E, int* s, int* d) {
    if (e < E) { *s = ei[e]; *d = ei[E + e]; }
    else       { *s = (int)(e - E); *d = (int)(e - E); }
}

__global__ void edge_max(const int* __restrict__ ei, int E, int Nn,
                         const float* __restrict__ ss, const float* __restrict__ sd,
                         float* __restrict__ m, int H)
{
    long long i = (long long)blockIdx.x * blockDim.x + threadIdx.x;
    long long tot = (long long)(E + Nn) * H;
    if (i >= tot) return;
    long long e = i / H; int hh = (int)(i % H);
    int s, d; edge_sd(ei, e, E, &s, &d);
    float v = ss[(long long)s * H + hh] + sd[(long long)d * H + hh];
    v = v > 0.0f ? v : 0.2f * v;                       // leaky_relu 0.2
    atomicMaxF(&m[(long long)d * H + hh], v);
}

__global__ void edge_expsum(const int* __restrict__ ei, int E, int Nn,
                            const float* __restrict__ ss, const float* __restrict__ sd,
                            const float* __restrict__ m, float* __restrict__ den, int H)
{
    long long i = (long long)blockIdx.x * blockDim.x + threadIdx.x;
    long long tot = (long long)(E + Nn) * H;
    if (i >= tot) return;
    long long e = i / H; int hh = (int)(i % H);
    int s, d; edge_sd(ei, e, E, &s, &d);
    float v = ss[(long long)s * H + hh] + sd[(long long)d * H + hh];
    v = v > 0.0f ? v : 0.2f * v;
    float ex = expf(v - m[(long long)d * H + hh]);
    atomicAdd(&den[(long long)d * H + hh], ex);
}

__global__ __launch_bounds__(256) void edge_agg(
    const int* __restrict__ ei, int E, int Nn,
    const float* __restrict__ ss, const float* __restrict__ sd,
    const float* __restrict__ m, const float* __restrict__ den,
    const float* __restrict__ h, float* __restrict__ agg, int H, int C)
{
    long long e = blockIdx.x;
    int s, d; edge_sd(ei, e, E, &s, &d);
    __shared__ float alpha[16];
    int tid = threadIdx.x;
    if (tid < H) {
        float v = ss[(long long)s * H + tid] + sd[(long long)d * H + tid];
        v = v > 0.0f ? v : 0.2f * v;
        alpha[tid] = expf(v - m[(long long)d * H + tid]) / den[(long long)d * H + tid];
    }
    __syncthreads();
    int HCt = H * C;
    const float* hs = h   + (long long)s * HCt;
    float*       ag = agg + (long long)d * HCt;
    for (int j = tid; j < HCt; j += blockDim.x)
        atomicAdd(&ag[j], hs[j] * alpha[j / C]);
}

__global__ void bias_relu(float* __restrict__ p, const float* __restrict__ b,
                          long long n, int F)
{
    long long i = (long long)blockIdx.x * blockDim.x + threadIdx.x;
    long long stride = (long long)gridDim.x * blockDim.x;
    for (; i < n; i += stride) p[i] = fmaxf(p[i] + b[(int)(i % F)], 0.0f);
}

__global__ void pool_max(const float* __restrict__ h, const int* __restrict__ batch,
                         float* __restrict__ g, int Nn, int C)
{
    long long i = (long long)blockIdx.x * blockDim.x + threadIdx.x;
    if (i >= (long long)Nn * C) return;
    int n = (int)(i / C), c = (int)(i % C);
    atomicMaxF(&g[(long long)batch[n] * C + c], h[i]);
}

// ---------------- protein branch: TDM-gathered embedding + conv1d ----------------
// TDM gather mode: one descriptor gathers 16 rows of emb (row width 128 f32) into LDS,
// row indices packed 16-bit in D# groups 2/3. 8 descriptors cover the 128-token sequence.
__device__ __forceinline__ void tdm_gather16(unsigned lds_addr, unsigned long long gaddr,
                                             const int* __restrict__ idx16)
{
    u32x4 g0;
    g0[0] = 0x80000001u;                               // gather_mode=1, 16b idx, count=1
    g0[1] = lds_addr;                                  // LDS dest byte address
    g0[2] = (unsigned)gaddr;                           // global_addr[31:0]
    g0[3] = ((unsigned)(gaddr >> 32) & 0x01FFFFFFu) | (2u << 30);   // addr hi | type=2
    i32x8 g1;
    g1[0] = 0x00020000;                                // data_size=4B, no multicast
    g1[1] = (EMBD << 16);                              // tensor_dim0 = 128
    g1[2] = (VOC << 16);                               // tensor_dim1 = 26 (rows, OOB)
    g1[3] = (EMBD << 16);                              // tile_dim0  = 128
    g1[4] = 16;                                        // tile_dim1  = #indices
    g1[5] = EMBD;                                      // tensor_dim0_stride = 128
    g1[6] = 0;
    g1[7] = 0;
    i32x4 g2, g3;
    #pragma unroll
    for (int j = 0; j < 4; ++j)
        g2[j] = (idx16[2 * j] & 0xFFFF) | (idx16[2 * j + 1] << 16);
    #pragma unroll
    for (int j = 0; j < 4; ++j)
        g3[j] = (idx16[8 + 2 * j] & 0xFFFF) | (idx16[8 + 2 * j + 1] << 16);
#if __has_include(<hip/amd_detail/amd_gfx1250_TDM.h>)
    i32x8 gz = {0, 0, 0, 0, 0, 0, 0, 0};
    __builtin_amdgcn_tensor_load_to_lds(g0, g1, g2, g3, gz, 0);
#else
    __builtin_amdgcn_tensor_load_to_lds(g0, g1, g2, g3, 0);
#endif
}

__global__ __launch_bounds__(128) void conv1d_tdm(
    const int* __restrict__ target, const float* __restrict__ emb,
    const float* __restrict__ W, const float* __restrict__ bias, float* __restrict__ out)
{
    __shared__ float se[SEQ * EMBD];                   // 64 KB: e[b] = emb[target[b,:]]
    const int b = blockIdx.x;
    if (threadIdx.x < 32) {                            // one wave issues the DMA
        unsigned lds_base = (unsigned)(size_t)(void*)se;
        unsigned long long ga = (unsigned long long)(size_t)emb;
        int idxbuf[16];
        for (int g = 0; g < SEQ / 16; ++g) {
            const int* tp = target + b * SEQ + g * 16;
            #pragma unroll
            for (int j = 0; j < 16; ++j) idxbuf[j] = tp[j];
            tdm_gather16(lds_base + g * 16 * EMBD * 4, ga, idxbuf);
        }
        __builtin_amdgcn_s_wait_tensorcnt(0);
    }
    __syncthreads();

    for (int o = threadIdx.x; o < NF * CONV_L; o += blockDim.x) {
        int f = o / CONV_L, l = o % CONV_L;
        const float* wf = W + f * SEQ * KCONV;
        float sum = 0.0f;
        for (int i = 0; i < SEQ; ++i) {                // "channels" = sequence positions
            const float* er = se + i * EMBD + l;
            const float* wr = wf + i * KCONV;
            #pragma unroll
            for (int k = 0; k < KCONV; ++k) sum += er[k] * wr[k];
        }
        sum += bias[f];
        out[(long long)b * (NF * CONV_L) + o] = fmaxf(sum, 0.0f);
    }
}

// ---------------- final head ----------------
__global__ void final_out(const float* __restrict__ f2, const float* __restrict__ Wo,
                          const float* __restrict__ bo, float* __restrict__ out)
{
    int r = blockIdx.x * blockDim.x + threadIdx.x;
    if (r >= BATCH_B) return;
    float s = 0.0f;
    for (int j = 0; j < 2 * OUTD; ++j) s += f2[r * 2 * OUTD + j] * Wo[j];
    out[r] = s + bo[0];
}

// ---------------- launch ----------------
extern "C" void kernel_launch(void* const* d_in, const int* in_sizes, int n_in,
                              void* d_out, int out_size, void* d_ws, size_t ws_size,
                              hipStream_t stream)
{
    const float* x       = (const float*)d_in[0];
    const int*   ei      = (const int*)  d_in[1];
    const int*   batch   = (const int*)  d_in[2];
    const int*   target  = (const int*)  d_in[3];
    const float* W1      = (const float*)d_in[4];
    const float* a_src1  = (const float*)d_in[5];
    const float* a_dst1  = (const float*)d_in[6];
    const float* b1      = (const float*)d_in[7];
    const float* W2      = (const float*)d_in[8];
    const float* a_src2  = (const float*)d_in[9];
    const float* a_dst2  = (const float*)d_in[10];
    const float* b2      = (const float*)d_in[11];
    const float* Wg      = (const float*)d_in[12];
    const float* bg      = (const float*)d_in[13];
    const float* emb     = (const float*)d_in[14];
    const float* convW   = (const float*)d_in[15];
    const float* convB   = (const float*)d_in[16];
    const float* Wxt     = (const float*)d_in[17];
    const float* bxt     = (const float*)d_in[18];
    const float* Wf1     = (const float*)d_in[19];
    const float* bf1     = (const float*)d_in[20];
    const float* Wf2     = (const float*)d_in[21];
    const float* bf2     = (const float*)d_in[22];
    const float* Wo      = (const float*)d_in[23];
    const float* bo      = (const float*)d_in[24];
    float* out = (float*)d_out;

    // ---- workspace layout (float elements) ----
    float* ws = (float*)d_ws;
    float* h1     = ws;                                   // 50000*780
    float* agg1   = h1     + (long long)N_NODES * HC1;    // 50000*780
    float* ss1    = agg1   + (long long)N_NODES * HC1;    // 50000*10
    float* sd1    = ss1    + (long long)N_NODES * HEADS;
    float* m1     = sd1    + (long long)N_NODES * HEADS;
    float* den1   = m1     + (long long)N_NODES * HEADS;
    float* h2     = den1   + (long long)N_NODES * HEADS;  // 50000*128
    float* agg2   = h2     + (long long)N_NODES * OUTD;   // 50000*128
    float* ss2    = agg2   + (long long)N_NODES * OUTD;   // 50000
    float* sd2    = ss2    + N_NODES;
    float* m2     = sd2    + N_NODES;
    float* den2   = m2     + N_NODES;
    float* convc  = den2   + N_NODES;                     // 256*3872
    float* xc     = convc  + (long long)BATCH_B * XTK;    // 256*256 (concat)
    float* f1     = xc     + (long long)BATCH_B * 2 * OUTD;     // 256*1024
    float* f2     = f1     + (long long)BATCH_B * 1024;   // 256*256
    float* gpool  = f2     + (long long)BATCH_B * 2 * OUTD;     // 256*128

    const float NEG_INF = -INFINITY;
    const int E2 = N_EDGES + N_NODES;   // edges incl. self loops

    // ---- init accumulators ----
    fill_f32<<<4096, 256, 0, stream>>>(m1,   NEG_INF, (long long)N_NODES * HEADS);
    fill_f32<<<4096, 256, 0, stream>>>(den1, 0.0f,    (long long)N_NODES * HEADS);
    fill_f32<<<4096, 256, 0, stream>>>(agg1, 0.0f,    (long long)N_NODES * HC1);
    fill_f32<<<4096, 256, 0, stream>>>(m2,   NEG_INF, (long long)N_NODES);
    fill_f32<<<4096, 256, 0, stream>>>(den2, 0.0f,    (long long)N_NODES);
    fill_f32<<<4096, 256, 0, stream>>>(agg2, 0.0f,    (long long)N_NODES * OUTD);
    fill_f32<<<256,  256, 0, stream>>>(gpool, NEG_INF, (long long)BATCH_B * OUTD);

    // ---- GAT layer 1 ----
    {   // h1 = x @ W1
        dim3 g((HC1 + 63) / 64, (N_NODES + 63) / 64);
        gemm_wmma_f16<<<g, 128, 0, stream>>>(x, W1, h1, nullptr,
            N_NODES, HC1, FXD, FXD, HC1, HC1, 0, 0);
    }
    {
        long long t = (long long)N_NODES * HEADS;
        gat_scores<<<(unsigned)((t + 255) / 256), 256, 0, stream>>>(h1, a_src1, a_dst1, ss1, sd1, N_NODES, HEADS, FXD);
        long long te = (long long)E2 * HEADS;
        edge_max<<<(unsigned)((te + 255) / 256), 256, 0, stream>>>(ei, N_EDGES, N_NODES, ss1, sd1, m1, HEADS);
        edge_expsum<<<(unsigned)((te + 255) / 256), 256, 0, stream>>>(ei, N_EDGES, N_NODES, ss1, sd1, m1, den1, HEADS);
        edge_agg<<<E2, 256, 0, stream>>>(ei, N_EDGES, N_NODES, ss1, sd1, m1, den1, h1, agg1, HEADS, FXD);
        bias_relu<<<4096, 256, 0, stream>>>(agg1, b1, (long long)N_NODES * HC1, HC1);
    }

    // ---- GAT layer 2 ----
    {   // h2 = agg1 @ W2
        dim3 g((OUTD + 63) / 64, (N_NODES + 63) / 64);
        gemm_wmma_f16<<<g, 128, 0, stream>>>(agg1, W2, h2, nullptr,
            N_NODES, OUTD, HC1, HC1, OUTD, OUTD, 0, 0);
    }
    {
        long long t = (long long)N_NODES;
        gat_scores<<<(unsigned)((t + 255) / 256), 256, 0, stream>>>(h2, a_src2, a_dst2, ss2, sd2, N_NODES, 1, OUTD);
        long long te = (long long)E2;
        edge_max<<<(unsigned)((te + 255) / 256), 256, 0, stream>>>(ei, N_EDGES, N_NODES, ss2, sd2, m2, 1);
        edge_expsum<<<(unsigned)((te + 255) / 256), 256, 0, stream>>>(ei, N_EDGES, N_NODES, ss2, sd2, m2, den2, 1);
        edge_agg<<<E2, 256, 0, stream>>>(ei, N_EDGES, N_NODES, ss2, sd2, m2, den2, h2, agg2, 1, OUTD);
        bias_relu<<<4096, 256, 0, stream>>>(agg2, b2, (long long)N_NODES * OUTD, OUTD);
    }

    // ---- global max pool + graph MLP into xc[:, :128] ----
    {
        long long t = (long long)N_NODES * OUTD;
        pool_max<<<(unsigned)((t + 255) / 256), 256, 0, stream>>>(agg2, batch, gpool, N_NODES, OUTD);
        dim3 g((OUTD + 63) / 64, (BATCH_B + 63) / 64);
        gemm_wmma_f16<<<g, 128, 0, stream>>>(gpool, Wg, xc, bg,
            BATCH_B, OUTD, OUTD, OUTD, OUTD, 2 * OUTD, 0, 1);    // relu
    }

    // ---- protein branch into xc[:, 128:] ----
    {
        conv1d_tdm<<<BATCH_B, 128, 0, stream>>>(target, emb, convW, convB, convc);
        dim3 g((OUTD + 63) / 64, (BATCH_B + 63) / 64);
        gemm_wmma_f16<<<g, 128, 0, stream>>>(convc, Wxt, xc, bxt,
            BATCH_B, OUTD, XTK, XTK, OUTD, 2 * OUTD, OUTD, 0);   // no relu on xt
    }

    // ---- fusion head ----
    {
        dim3 g1((1024 + 63) / 64, (BATCH_B + 63) / 64);
        gemm_wmma_f16<<<g1, 128, 0, stream>>>(xc, Wf1, f1, bf1,
            BATCH_B, 1024, 2 * OUTD, 2 * OUTD, 1024, 1024, 0, 1);
        dim3 g2((2 * OUTD + 63) / 64, (BATCH_B + 63) / 64);
        gemm_wmma_f16<<<g2, 128, 0, stream>>>(f1, Wf2, f2, bf2,
            BATCH_B, 2 * OUTD, 1024, 1024, 2 * OUTD, 2 * OUTD, 0, 1);
        final_out<<<(BATCH_B + 127) / 128, 128, 0, stream>>>(f2, Wo, bo, out);
    }
}
